// STULayer_56702158242312
// MI455X (gfx1250) — matrix operand order repaired and checked by reading, not compiled, not run on hardware
//
#include <hip/hip_runtime.h>

// HSTU delta attention for MI455X (gfx1250), wave32 + WMMA f16.
//   S^T = Ktile * Q^T   (v_wmma_f32_16x16x32_f16, K-dim = d, 4 chained)
//   P   = mask * silu(alpha*S) / N     (fast exp2/rcp, masking folded in)
//   O^T += V^T * P^T    (v_wmma_f32_16x16x32_f16, K-dim = n, chained over tiles)
// Shapes hardcoded to the harness: H=8, D=128, DELTA=256, N=2048.

typedef __attribute__((ext_vector_type(16))) _Float16 v16h;
typedef __attribute__((ext_vector_type(8)))  _Float16 v8h;
typedef __attribute__((ext_vector_type(2)))  _Float16 v2h;
typedef __attribute__((ext_vector_type(8)))  float    v8f;

#define H_      8
#define D_      128
#define HD_     (H_ * D_)
#define DELTA_  256
#define NMAX_   2048
#define NT_     64
#define KT_PITCH 136   // halfs per K-tile row   (64 rows x 128 d), 272B: 16B aligned, conflict-friendly
#define VT_PITCH 72    // halfs per V^T-tile row (128 d  x 64 n), 144B

#define LOG2E_F 1.44269504088896340736f

__device__ __forceinline__ v16h lds_frag(const _Float16* p0, const _Float16* p1) {
  v8h lo = *(const v8h*)p0;   // ds_load_b128
  v8h hi = *(const v8h*)p1;   // ds_load_b128
  v16h r;
#pragma unroll
  for (int i = 0; i < 8; ++i) { r[i] = lo[i]; r[i + 8] = hi[i]; }
  return r;
}

__device__ __forceinline__ unsigned pack_f16(float a, float b) {
  return __builtin_bit_cast(unsigned, __builtin_amdgcn_cvt_pkrtz(a, b)); // v_cvt_pk_rtz_f16_f32
}

__global__ __launch_bounds__(256)
void hstu_delta_wmma(const float* __restrict__ q,
                     const float* __restrict__ kg,
                     const float* __restrict__ vg,
                     const int*   __restrict__ seq_offsets,
                     const float* __restrict__ alpha_p,
                     float* __restrict__ out)
{
  __shared__ __align__(16) _Float16 Kt[NT_ * KT_PITCH]; // [n][d]  row-major f16
  __shared__ __align__(16) _Float16 Vt[D_ * VT_PITCH];  // [d][n]  transposed f16

  const int bid  = blockIdx.x;          // ((b*H)+h)*2 + half
  const int half = bid & 1;
  const int h    = (bid >> 1) & (H_ - 1);
  const int b    = bid >> 4;

  const int tid  = threadIdx.x;
  const int wave = tid >> 5;
  const int lane = tid & 31;
  const int ml   = lane & 15;
  const bool hif = lane >= 16;

  const int   off   = seq_offsets[b];
  const int   len   = seq_offsets[b + 1] - off;
  const float alpha = alpha_p[0];
  const float c_sig = -alpha * LOG2E_F;              // s -> exponent for exp2
  const float c_out = alpha * (1.0f / (float)NMAX_); // s -> alpha*s/N

  const int mq      = half * 128 + wave * 16;      // m-tile start in [0,256)
  const int row_lim = (NMAX_ - DELTA_) + mq + ml;  // causal col cap for this lane's query
  const int lim     = (row_lim < len - 1) ? row_lim : (len - 1); // combined cap
  const int rl_wave = (NMAX_ - DELTA_) + mq + 15;  // wave-uniform cap

  const long qrow = (long)b * DELTA_ + mq + ml;
  const float* qp = q + qrow * HD_ + h * D_;

  // ---- persistent Q^T B-fragments: 4 x (32 d-slice), 16 consecutive f32 per lane ----
  v16h qb[4];
#pragma unroll
  for (int j = 0; j < 4; ++j) {
    const float4* p4 = (const float4*)(qp + j * 32 + (hif ? 16 : 0));
    v16h t;
#pragma unroll
    for (int i = 0; i < 4; ++i) {
      float4 f = p4[i];
      t[4*i+0] = (_Float16)f.x; t[4*i+1] = (_Float16)f.y;
      t[4*i+2] = (_Float16)f.z; t[4*i+3] = (_Float16)f.w;
    }
    qb[j] = t;
  }

  // ---- O^T accumulators: 8 d-fragments of 16x16 f32 ----
  v8f o[8];
#pragma unroll
  for (int f = 0; f < 8; ++f)
#pragma unroll
    for (int r = 0; r < 8; ++r) o[f][r] = 0.0f;

  int n_end = (NMAX_ - DELTA_) + half * 128 + 128;
  if (len < n_end) n_end = len;

  // per-thread loop-invariant staging addresses
  _Float16* const ks = &Kt[(tid >> 6) * KT_PITCH + 2 * (tid & 63)];
  _Float16* const vs = &Vt[(tid & 127) * VT_PITCH + 2 * (tid >> 7)];
  const long kg_inv = ((long)off + (tid >> 6)) * HD_ + h * D_ + 2 * (tid & 63);
  const long vg_inv = ((long)off + 2 * (tid >> 7)) * HD_ + h * D_ + (tid & 127);

  for (int n0 = 0; n0 < n_end; n0 += NT_) {
    if (n0 + NT_ <= len) {
      // ---------- fast path: full tile, base + immediate-offset loads ----------
      const float* kb = kg + kg_inv + (long)n0 * HD_;
      const float* vb = vg + vg_inv + (long)n0 * HD_;
#pragma unroll
      for (int i = 0; i < 16; ++i) {                 // K rows advance by 4 per slot
        float2 kk = *(const float2*)(kb + (long)i * 4 * HD_);
        ks[i * 4 * KT_PITCH]     = (_Float16)kk.x;
        ks[i * 4 * KT_PITCH + 1] = (_Float16)kk.y;
      }
#pragma unroll
      for (int i = 0; i < 16; ++i) {                 // V rows advance by 4 per slot
        float fa = vb[(long)i * 4 * HD_];
        float fb = vb[(long)i * 4 * HD_ + HD_];
        vs[i * 4]     = (_Float16)fa;
        vs[i * 4 + 1] = (_Float16)fb;
      }
    } else {
      // ---------- tail path: clamped rows (masked later anyway) ----------
#pragma unroll
      for (int i = 0; i < 16; ++i) {
        int s  = tid + i * 256;
        int r  = s >> 6;
        int c2 = s & 63;
        int nr = n0 + r; if (nr >= len) nr = len - 1;
        const float2 kk = *(const float2*)(kg + (long)(off + nr) * HD_ + h * D_ + 2 * c2);
        Kt[r * KT_PITCH + 2 * c2]     = (_Float16)kk.x;
        Kt[r * KT_PITCH + 2 * c2 + 1] = (_Float16)kk.y;
      }
#pragma unroll
      for (int i = 0; i < 16; ++i) {
        int s  = tid + i * 256;
        int d  = s & 127;
        int n2 = s >> 7;
        int na = n0 + 2 * n2, nb = na + 1;
        if (na >= len) na = len - 1;
        if (nb >= len) nb = len - 1;
        float fa = vg[(long)(off + na) * HD_ + h * D_ + d];
        float fb = vg[(long)(off + nb) * HD_ + h * D_ + d];
        Vt[d * VT_PITCH + 2 * n2]     = (_Float16)fa;
        Vt[d * VT_PITCH + 2 * n2 + 1] = (_Float16)fb;
      }
    }
    __syncthreads();

    if (n0 <= rl_wave) {               // wave-uniform: EXEC stays all-ones
#pragma unroll
      for (int nn = 0; nn < NT_; nn += 32) {
        // ---- gemm1: two S^T fragments (key rows nn..nn+15, nn+16..nn+31) ----
        v8f S[2];
#pragma unroll
        for (int g = 0; g < 2; ++g) {
          v8f sf;
#pragma unroll
          for (int r = 0; r < 8; ++r) sf[r] = 0.0f;
          const int krow = (nn + g * 16 + ml) * KT_PITCH + (hif ? 16 : 0);
#pragma unroll
          for (int c = 0; c < 4; ++c) {
            v16h ka = lds_frag(&Kt[krow + 32 * c], &Kt[krow + 32 * c + 8]);
            sf = __builtin_amdgcn_wmma_f32_16x16x32_f16(false, ka, false, qb[c],
                                                        (short)0, sf, false, false);
          }
          // silu + combined causal/length mask (element (n_local=r+hif*8, m=ml))
          const int rr = lim - (n0 + nn + g * 16 + (hif ? 8 : 0));
#pragma unroll
          for (int r = 0; r < 8; ++r) {
            float s = sf[r];
            float e = __builtin_amdgcn_exp2f(s * c_sig);           // exp(-alpha*s)
            float p = s * c_out * __builtin_amdgcn_rcpf(1.0f + e); // silu(alpha*s)/N
            sf[r] = (r <= rr) ? p : 0.0f;
          }
          S[g] = sf;
        }
        // ---- convert S^T C-layout pair -> P^T B-fragment (f16) ----
        v16h pb;
#pragma unroll
        for (int j = 0; j < 8; ++j) {
          const int r0 = (j & 3) * 2;
          unsigned ua = pack_f16(S[0][r0], S[0][r0 + 1]);
          unsigned ub = pack_f16(S[1][r0], S[1][r0 + 1]);
          unsigned res;
          if (j < 4) {                 // K=n' 0..7   : lo lanes same-lane F0, hi lanes F1 from lane^16
            unsigned sb = (unsigned)__shfl_xor((int)ub, 16, 32);
            res = hif ? sb : ua;
          } else {                     // K=n' 8..15  : lo lanes F0 from lane^16, hi lanes same-lane F1
            unsigned sa = (unsigned)__shfl_xor((int)ua, 16, 32);
            res = hif ? ub : sa;
          }
          v2h r2 = __builtin_bit_cast(v2h, res);
          pb[2 * j] = r2[0]; pb[2 * j + 1] = r2[1];
        }
        // ---- gemm2: O^T += V^T * P^T over 8 d-fragments ----
#pragma unroll
        for (int f = 0; f < 8; ++f) {
          const int vrow = (16 * f + ml) * VT_PITCH + nn + (hif ? 16 : 0);
          v16h va = lds_frag(&Vt[vrow], &Vt[vrow + 8]);
          o[f] = __builtin_amdgcn_wmma_f32_16x16x32_f16(false, va, false, pb,
                                                        (short)0, o[f], false, false);
        }
      }
    }
    __syncthreads();
  }

  // ---- epilogue: O^T C-layout -> out[qrow][h][d], contiguous 8-f32 runs per lane ----
  float* op = out + qrow * HD_ + h * D_;
#pragma unroll
  for (int f = 0; f < 8; ++f) {
    float4 s0, s1;
    s0.x = o[f][0]; s0.y = o[f][1]; s0.z = o[f][2]; s0.w = o[f][3];
    s1.x = o[f][4]; s1.y = o[f][5]; s1.z = o[f][6]; s1.w = o[f][7];
    float* dst = op + 16 * f + (hif ? 8 : 0);
    *(float4*)(dst)     = s0;   // global_store_b128
    *(float4*)(dst + 4) = s1;   // global_store_b128
  }
}

extern "C" void kernel_launch(void* const* d_in, const int* in_sizes, int n_in,
                              void* d_out, int out_size, void* d_ws, size_t ws_size,
                              hipStream_t stream) {
  const float* dq    = (const float*)d_in[0];
  const float* k     = (const float*)d_in[1];
  const float* v     = (const float*)d_in[2];
  const int*   offs  = (const int*)d_in[3];
  const float* alpha = (const float*)d_in[4];
  (void)n_in; (void)out_size; (void)d_ws; (void)ws_size;

  const int B    = in_sizes[3] - 1;       // seq_offsets has B+1 entries
  const int grid = B * H_ * 2;            // (b, h, half) blocks
  hstu_delta_wmma<<<grid, 256, 0, stream>>>(dq, k, v, offs, alpha, (float*)d_out);
}
